// SelfAttention_5574867550217
// MI455X (gfx1250) — matrix operand order repaired and checked by reading, not compiled, not run on hardware
//
#include <hip/hip_runtime.h>
#include <hip/hip_bf16.h>

typedef __attribute__((ext_vector_type(16))) _Float16 v16h;
typedef __attribute__((ext_vector_type(8)))  _Float16 v8h;
typedef __attribute__((ext_vector_type(8)))  float    v8f;
typedef __attribute__((ext_vector_type(4)))  unsigned int u32x4;
typedef __attribute__((ext_vector_type(8)))  int i32x8;
typedef __attribute__((ext_vector_type(4)))  int i32x4;

#define B_ 2
#define L_ 2048
#define E_ 1024
#define H_ 16
#define D_ 64

__device__ __forceinline__ v16h cat8(v8h lo, v8h hi) {
    return __builtin_shufflevector(lo, hi, 0,1,2,3,4,5,6,7,8,9,10,11,12,13,14,15);
}

// ---------------------------------------------------------------------------
// Tensor Data Mover: 2D tile (rows x cols of f16) global -> LDS.
// Descriptor bitfields per CDNA5 ISA ch.8 (group0: count/lds/global/type,
// group1: data_size + tensor dims + tile dims + dim0 stride).
// ---------------------------------------------------------------------------
__device__ __forceinline__ void tdm_load_2d_f16(const void* gaddr, unsigned lds_byte_addr,
                                                unsigned tensor_d0, unsigned tensor_d1,
                                                unsigned tile_d0, unsigned tile_d1,
                                                unsigned stride0_elems)
{
    unsigned long long ga = (unsigned long long)gaddr;
    u32x4 g0;
    g0[0] = 1u;                                            // count=1 (valid D#)
    g0[1] = lds_byte_addr;                                 // LDS byte address
    g0[2] = (unsigned)(ga & 0xFFFFFFFFu);                  // global_addr[31:0]
    g0[3] = (unsigned)((ga >> 32) & 0x1FFFFFFu)            // global_addr[56:32]
          | (2u << 30);                                    // type=2 ("image")
    i32x8 g1;
    g1[0] = (int)(1u << 16);                               // data_size=1 (2B), wg_mask=0
    g1[1] = (int)((tensor_d0 & 0xFFFFu) << 16);            // tensor_dim0[15:0]
    g1[2] = (int)((tensor_d0 >> 16) | ((tensor_d1 & 0xFFFFu) << 16));
    g1[3] = (int)((tensor_d1 >> 16) | (tile_d0 << 16));    // tile_dim0
    g1[4] = (int)(tile_d1 & 0xFFFFu);                      // tile_dim1, tile_dim2=0
    g1[5] = (int)stride0_elems;                            // tensor_dim0_stride[31:0]
    g1[6] = 0;                                             // stride0 hi / stride1 lo
    g1[7] = 0;
    i32x4 z4 = {0, 0, 0, 0};
#if defined(__clang_major__) && (__clang_major__ >= 23)
    i32x8 z8 = {0, 0, 0, 0, 0, 0, 0, 0};
    __builtin_amdgcn_tensor_load_to_lds(g0, g1, z4, z4, z8, 0);
#else
    __builtin_amdgcn_tensor_load_to_lds(g0, g1, z4, z4, 0);
#endif
}

// ---------------------------------------------------------------------------
// f32 -> f16 conversion (n divisible by 1024)
// ---------------------------------------------------------------------------
__global__ __launch_bounds__(256)
void cvt_f32_to_f16(const float* __restrict__ src, _Float16* __restrict__ dst, int n) {
    int i = (blockIdx.x * blockDim.x + threadIdx.x) * 4;
    if (i + 3 < n) {
        float4 v = *(const float4*)(src + i);
        dst[i + 0] = (_Float16)v.x;
        dst[i + 1] = (_Float16)v.y;
        dst[i + 2] = (_Float16)v.z;
        dst[i + 3] = (_Float16)v.w;
    }
}

// ---------------------------------------------------------------------------
// Unified WMMA GEMM with TDM-staged, double-buffered LDS tiles.
//   C[M,N] = (A[M,K] @ W[N,K]^T + bias[N]) * scale
// Block tile 64x256 (8 waves; wave = 32x64), K-step 32.
// Wave 0 issues two tensor_load_to_lds per step into the "next" buffer while
// all waves compute from the "current" buffer; s_wait_tensorcnt + barrier.
// mode 0: f16 store [B,H,L,D] (Q/K)   mode 1: f16 store [B,H,D,L] (V^T)
// mode 2: f32 store row-major [M,N]   (final projection)
// ---------------------------------------------------------------------------
#define KSTEP 32
#define A_TILE_H (64 * KSTEP)                 // 2048 halves
#define W_TILE_H (256 * KSTEP)                // 8192 halves
#define BUF_H    (A_TILE_H + W_TILE_H)        // 10240 halves per buffer

__global__ __launch_bounds__(256)
void wmma_gemm_tdm(const _Float16* __restrict__ A, const _Float16* __restrict__ W,
                   const float* __restrict__ bias,
                   _Float16* __restrict__ out16, float* __restrict__ out32,
                   int M, int N, int K, float scale, int mode)
{
    __shared__ _Float16 smem[2 * BUF_H];      // 40 KB, double buffered

    const int lane = threadIdx.x & 31;
    const int hf   = lane >> 4;               // K-sub-chunk select (WMMA layout)
    const int l16  = lane & 15;
    const int wv   = threadIdx.x >> 5;
    const int wvM  = wv >> 2;                 // 0..1  -> 32-row strip
    const int wvN  = wv & 3;                  // 0..3  -> 64-col strip
    const int m0b  = blockIdx.x * 64;
    const int n0b  = blockIdx.y * 256;

    const unsigned smemBase = (unsigned)(unsigned long long)(const void*)smem;

    v8f acc[2][4] = {};

    const int nk = K / KSTEP;
    // prologue: stage K-step 0 into buffer 0
    if (wv == 0) {
        tdm_load_2d_f16((const char*)A + (size_t)m0b * K * 2, smemBase,
                        (unsigned)K, (unsigned)(M - m0b), KSTEP, 64, (unsigned)K);
        tdm_load_2d_f16((const char*)W + (size_t)n0b * K * 2,
                        smemBase + A_TILE_H * 2,
                        (unsigned)K, (unsigned)(N - n0b), KSTEP, 256, (unsigned)K);
        __builtin_amdgcn_s_wait_tensorcnt(0);
    }
    __syncthreads();

    for (int i = 0; i < nk; ++i) {
        const int cur = i & 1;
        const int nxt = cur ^ 1;
        // kick off DMA for the next K-step while we compute this one
        if (wv == 0 && (i + 1) < nk) {
            const int k1 = (i + 1) * KSTEP;
            tdm_load_2d_f16((const char*)A + ((size_t)m0b * K + k1) * 2,
                            smemBase + (unsigned)(nxt * BUF_H) * 2,
                            (unsigned)K, (unsigned)(M - m0b), KSTEP, 64, (unsigned)K);
            tdm_load_2d_f16((const char*)W + ((size_t)n0b * K + k1) * 2,
                            smemBase + (unsigned)(nxt * BUF_H + A_TILE_H) * 2,
                            (unsigned)K, (unsigned)(N - n0b), KSTEP, 256, (unsigned)K);
        }

        const _Float16* aT = smem + cur * BUF_H;             // [64][32]
        const _Float16* wT = aT + A_TILE_H;                  // [256][32]

        // B fragments: column n = l16; lanes 0-15 K=0..15, 16-31 K=16..31
        v16h bf[4];
#pragma unroll
        for (int t = 0; t < 4; ++t) {
            const int wr = wvN * 64 + t * 16 + l16;
            bf[t] = *(const v16h*)(wT + wr * KSTEP + hf * 16);
        }
        // A fragments: lane holds row l16; K 0..15 / 16..31
        v16h af[2];
#pragma unroll
        for (int mi = 0; mi < 2; ++mi) {
            const int r = wvM * 32 + mi * 16 + l16;
            v8h alo = *(const v8h*)(aT + r * KSTEP + hf * 16);
            v8h ahi = *(const v8h*)(aT + r * KSTEP + hf * 16 + 8);
            af[mi] = cat8(alo, ahi);
        }
#pragma unroll
        for (int mi = 0; mi < 2; ++mi)
#pragma unroll
            for (int t = 0; t < 4; ++t)
                acc[mi][t] = __builtin_amdgcn_wmma_f32_16x16x32_f16(
                    false, af[mi], false, bf[t], (short)0, acc[mi][t], false, false);

        if (wv == 0 && (i + 1) < nk)
            __builtin_amdgcn_s_wait_tensorcnt(0);   // next buffer fully landed
        __syncthreads();                            // everyone done reading cur
    }

    // epilogue: bias + scale + store (C layout: VGPR j -> row j / j+8)
#pragma unroll
    for (int mi = 0; mi < 2; ++mi) {
#pragma unroll
        for (int t = 0; t < 4; ++t) {
#pragma unroll
            for (int j = 0; j < 8; ++j) {
                const int m = m0b + wvM * 32 + mi * 16 + j + hf * 8;
                const int n = n0b + wvN * 64 + t * 16 + l16;
                const float v = (acc[mi][t][j] + bias[n]) * scale;
                if (mode == 2) {
                    out32[(size_t)m * N + n] = v;
                } else {
                    const int b = m >> 11, l = m & (L_ - 1);
                    const int h = n >> 6,  d = n & (D_ - 1);
                    const size_t idx = (mode == 0)
                        ? ((size_t)(b * H_ + h) * L_ + l) * D_ + d
                        : ((size_t)(b * H_ + h) * D_ + d) * L_ + l;
                    out16[idx] = (_Float16)v;
                }
            }
        }
    }
}

// ---------------------------------------------------------------------------
// Flash attention: one wave per (b, h, 16-query tile); 32-key chunks.
// 4 WMMAs for S = Q K^T, online softmax (16-lane shfl reductions match the
// C-layout half-wave row striping), P reshaped C-layout -> A-layout via LDS,
// 4 WMMAs for O += P V. Latency hidden by 4096 independent workgroups.
// ---------------------------------------------------------------------------
__global__ __launch_bounds__(32)
void flash_attn(const _Float16* __restrict__ Q,   // [B,H,L,D] (pre-scaled D^-0.5)
                const _Float16* __restrict__ Kk,  // [B,H,L,D]
                const _Float16* __restrict__ Vt,  // [B,H,D,L]
                const float* __restrict__ mask,   // [B,1,L,L]
                _Float16* __restrict__ Oa)        // [B,L,H*D]
{
    const int lane = threadIdx.x & 31;
    const int hf   = lane >> 4;
    const int l16  = lane & 15;
    const int bid  = blockIdx.x;
    const int qt = bid & (L_ / 16 - 1);
    const int h  = (bid >> 7) & (H_ - 1);
    const int b  = bid >> 11;

    __shared__ _Float16 pLds[16 * 32];

    const _Float16* qbase = Q  + ((size_t)(b * H_ + h) * L_ + qt * 16) * D_;
    const _Float16* kbase = Kk + (size_t)(b * H_ + h) * L_ * D_;
    const _Float16* vbase = Vt + (size_t)(b * H_ + h) * D_ * L_;
    const float*    mbase = mask + ((size_t)b * L_ + qt * 16) * L_;

    v16h aq[2];
    {
        const _Float16* qrow = qbase + (size_t)l16 * D_;
#pragma unroll
        for (int ks = 0; ks < 2; ++ks) {
            v8h lo = *(const v8h*)(qrow + ks * 32 + hf * 16);
            v8h hi = *(const v8h*)(qrow + ks * 32 + hf * 16 + 8);
            aq[ks] = cat8(lo, hi);
        }
    }

    v8f o[4] = {};
    float mrun[8], lrun[8];
#pragma unroll
    for (int j = 0; j < 8; ++j) { mrun[j] = -1e30f; lrun[j] = 0.0f; }

    for (int kt = 0; kt < L_; kt += 32) {
        v8f s[2] = {};
#pragma unroll
        for (int t = 0; t < 2; ++t) {
#pragma unroll
            for (int ks = 0; ks < 2; ++ks) {
                const _Float16* krow =
                    kbase + (size_t)(kt + t * 16 + l16) * D_ + ks * 32 + hf * 16;
                v16h bf = *(const v16h*)krow;
                s[t] = __builtin_amdgcn_wmma_f32_16x16x32_f16(
                    false, aq[ks], false, bf, (short)0, s[t], false, false);
            }
#pragma unroll
            for (int j = 0; j < 8; ++j) {
                int qr = j + hf * 8;
                s[t][j] += mbase[(size_t)qr * L_ + kt + t * 16 + l16];
            }
        }

        // online softmax per row (row j / j+8 per half-wave)
#pragma unroll
        for (int j = 0; j < 8; ++j) {
            float mx = fmaxf(s[0][j], s[1][j]);
#pragma unroll
            for (int sm = 1; sm < 16; sm <<= 1)
                mx = fmaxf(mx, __shfl_xor(mx, sm, 32));
            float mnew = fmaxf(mrun[j], mx);
            float corr = __expf(mrun[j] - mnew);
            float p0 = __expf(s[0][j] - mnew);
            float p1 = __expf(s[1][j] - mnew);
            s[0][j] = p0; s[1][j] = p1;
            float rs = p0 + p1;
#pragma unroll
            for (int sm = 1; sm < 16; sm <<= 1)
                rs += __shfl_xor(rs, sm, 32);
            lrun[j] = lrun[j] * corr + rs;
            mrun[j] = mnew;
#pragma unroll
            for (int t = 0; t < 4; ++t)
                o[t][j] *= corr;
        }

        // P: C-layout -> LDS [16 x 32] -> A-layout fragment
#pragma unroll
        for (int t = 0; t < 2; ++t)
#pragma unroll
            for (int j = 0; j < 8; ++j)
                pLds[(j + hf * 8) * 32 + t * 16 + l16] = (_Float16)s[t][j];
        __syncthreads();
        v16h pf;
        {
            v8h lo = *(const v8h*)&pLds[l16 * 32 + hf * 16];
            v8h hi = *(const v8h*)&pLds[l16 * 32 + hf * 16 + 8];
            pf = cat8(lo, hi);
        }
        __syncthreads();

        // O += P V  (B operand contiguous thanks to transposed V)
#pragma unroll
        for (int t = 0; t < 4; ++t) {
            const _Float16* vrow = vbase + (size_t)(t * 16 + l16) * L_ + kt + hf * 16;
            v16h bf = *(const v16h*)vrow;
            o[t] = __builtin_amdgcn_wmma_f32_16x16x32_f16(
                false, pf, false, bf, (short)0, o[t], false, false);
        }
    }

#pragma unroll
    for (int t = 0; t < 4; ++t)
#pragma unroll
        for (int j = 0; j < 8; ++j) {
            int row = j + hf * 8;
            float v = o[t][j] / lrun[j];
            size_t idx = ((size_t)b * L_ + qt * 16 + row) * (H_ * D_)
                       + h * D_ + t * 16 + l16;
            Oa[idx] = (_Float16)v;
        }
}

// ---------------------------------------------------------------------------
extern "C" void kernel_launch(void* const* d_in, const int* in_sizes, int n_in,
                              void* d_out, int out_size, void* d_ws, size_t ws_size,
                              hipStream_t stream) {
    const float* x    = (const float*)d_in[0];
    const float* mask = (const float*)d_in[1];
    const float* Wq   = (const float*)d_in[2];
    const float* bq   = (const float*)d_in[3];
    const float* Wk   = (const float*)d_in[4];
    const float* bk   = (const float*)d_in[5];
    const float* Wv   = (const float*)d_in[6];
    const float* bv   = (const float*)d_in[7];
    const float* Wo   = (const float*)d_in[8];
    const float* bo   = (const float*)d_in[9];
    float* out = (float*)d_out;

    const size_t XSZ = (size_t)B_ * L_ * E_;    // 4,194,304
    const size_t WSZ = (size_t)E_ * E_;         // 1,048,576
    _Float16* x16  = (_Float16*)d_ws;
    _Float16* wq16 = x16  + XSZ;
    _Float16* wk16 = wq16 + WSZ;
    _Float16* wv16 = wk16 + WSZ;
    _Float16* wo16 = wv16 + WSZ;
    _Float16* q16  = wo16 + WSZ;                // [B,H,L,D]
    _Float16* k16  = q16  + XSZ;                // [B,H,L,D]
    _Float16* vt16 = k16  + XSZ;                // [B,H,D,L]
    _Float16* a16  = vt16 + XSZ;                // [B,L,H*D]

    // 1) f32 -> f16
    cvt_f32_to_f16<<<(int)(XSZ / 1024), 256, 0, stream>>>(x,  x16,  (int)XSZ);
    cvt_f32_to_f16<<<(int)(WSZ / 1024), 256, 0, stream>>>(Wq, wq16, (int)WSZ);
    cvt_f32_to_f16<<<(int)(WSZ / 1024), 256, 0, stream>>>(Wk, wk16, (int)WSZ);
    cvt_f32_to_f16<<<(int)(WSZ / 1024), 256, 0, stream>>>(Wv, wv16, (int)WSZ);
    cvt_f32_to_f16<<<(int)(WSZ / 1024), 256, 0, stream>>>(Wo, wo16, (int)WSZ);

    const int M = B_ * L_;                      // 4096
    const int N = H_ * D_;                      // 1024
    const int K = E_;                           // 1024
    dim3 gg(M / 64, N / 256);                   // (64, 4)

    // 2) Q/K/V projections (Q pre-scaled by D^-0.5; V stored transposed)
    wmma_gemm_tdm<<<gg, 256, 0, stream>>>(x16, wq16, bq, q16,  nullptr, M, N, K, 0.125f, 0);
    wmma_gemm_tdm<<<gg, 256, 0, stream>>>(x16, wk16, bk, k16,  nullptr, M, N, K, 1.0f,   0);
    wmma_gemm_tdm<<<gg, 256, 0, stream>>>(x16, wv16, bv, vt16, nullptr, M, N, K, 1.0f,   1);

    // 3) flash attention: B*H*(L/16) = 4096 single-wave workgroups
    flash_attn<<<B_ * H_ * (L_ / 16), 32, 0, stream>>>(q16, k16, vt16, mask, a16);

    // 4) output projection -> f32
    wmma_gemm_tdm<<<gg, 256, 0, stream>>>(a16, wo16, bo, nullptr, out, M, E_, N, 1.0f, 2);
}
